// CausalSelfAttention_32435593020150
// MI455X (gfx1250) — compile-verified, matrix operand order checked by reading, not run on hardware
//
#include <hip/hip_runtime.h>
#include <math.h>

typedef unsigned short u16t;
typedef __attribute__((ext_vector_type(16))) __bf16 v16bf;
typedef __attribute__((ext_vector_type(8)))  float  v8f;

#define B_  2
#define S_  2048
#define E_  1024
#define H_  16
#define D_  64
#define N3_ 3072
#define M_  (B_*S_)

#define STILE (128*32)   // one LDS staging tile: 128 rows x 32 k (bf16)

__device__ __forceinline__ u16t f2bf(float f) {
  unsigned int u = __float_as_uint(f);
  unsigned int r = (u + 0x7FFFu + ((u >> 16) & 1u)) >> 16;   // round-to-nearest-even
  return (u16t)r;
}

union FragU { v16bf bf; uint4 q[2]; };

// 16-element bf16 fragment slice: elements 0..7 at p, 8..15 at p+16
// (CDNA5 16-bit A/B fragment layout: K = kb+{0..7}, kb+{16..23}).
__device__ __forceinline__ v16bf load_frag(const u16t* p) {
  FragU f;
  f.q[0] = *(const uint4*)(p);
  f.q[1] = *(const uint4*)(p + 16);
  return f.bf;
}

// ---- gfx1250 async global->LDS copy (ASYNCcnt-tracked, no VGPR staging) ----
__device__ __forceinline__ void async_b128(void* lds, const void* g) {
  unsigned loff = (unsigned)(size_t)lds;        // LDS aperture: addr[31:0] is the LDS byte address
  asm volatile("global_load_async_to_lds_b128 %0, %1, off"
               :: "v"(loff), "v"(g) : "memory");
}
__device__ __forceinline__ void wait_async() {
  asm volatile("s_wait_asynccnt 0" ::: "memory");
}

// ---------------- conversion kernels ----------------

__global__ void k_cvt(const float* __restrict__ in, u16t* __restrict__ out, int n) {
  int i = blockIdx.x * blockDim.x + threadIdx.x;
  if (i < n) out[i] = f2bf(in[i]);
}

// in[K][N] (row-major) -> out[N][K] bf16 (so GEMM B-fragments are contiguous)
__global__ void k_cvt_t(const float* __restrict__ in, u16t* __restrict__ out, int K, int N) {
  int k = blockIdx.x * blockDim.x + threadIdx.x;
  int n = blockIdx.y;
  if (k < K) out[(size_t)n * K + k] = f2bf(in[(size_t)k * N + n]);
}

// ---------------- shared GEMM mainloop ----------------
// 256 threads = 8 waves. Block tile 128(M) x 128(N); wave tile 32x64 = 2x4 WMMA tiles.
// A: [M][KD] bf16 row-major. Bt: [N][KD] bf16 (B transposed).
// Double-buffered LDS fed by async global->LDS DMA.

template<int KD>
__device__ __forceinline__ void issue_stage(const u16t* __restrict__ A,
                                            const u16t* __restrict__ Bt,
                                            int m0, int n0, int k0,
                                            u16t* Asb, u16t* Bsb) {
  const int tid = threadIdx.x;
  #pragma unroll
  for (int it = 0; it < 2; ++it) {
    int idx = tid + it * 256;          // 0..511 -> 128 rows x 4 uint4
    int row = idx >> 2;
    int q   = (idx & 3) * 8;           // bf16 element offset within 32-wide k slice
    async_b128(&Asb[row * 32 + q], &A [(size_t)(m0 + row) * KD + k0 + q]);
    async_b128(&Bsb[row * 32 + q], &Bt[(size_t)(n0 + row) * KD + k0 + q]);
  }
}

template<int KD>
__device__ __forceinline__ void gemm_main(const u16t* __restrict__ A,
                                          const u16t* __restrict__ Bt,
                                          int m0, int n0,
                                          u16t* As, u16t* Bs, v8f acc[2][4]) {
  const int lane = threadIdx.x & 31;
  const int wid  = threadIdx.x >> 5;
  const int wm = (wid & 3) * 32;       // 4 waves along M
  const int wn = (wid >> 2) * 64;      // 2 waves along N
  const int lm = lane & 15;
  const int kb = (lane < 16) ? 0 : 8;

  issue_stage<KD>(A, Bt, m0, n0, 0, As, Bs);   // prime buffer 0
  int buf = 0;

  for (int k0 = 0; k0 < KD; k0 += 32) {
    wait_async();                      // this wave's DMA into buf done
    __syncthreads();                   // everyone's DMA into buf done
    if (k0 + 32 < KD)                  // prefetch next tile into other buffer (overlaps compute)
      issue_stage<KD>(A, Bt, m0, n0, k0 + 32, As + (buf ^ 1) * STILE, Bs + (buf ^ 1) * STILE);

    const u16t* Ab = As + buf * STILE;
    const u16t* Bb = Bs + buf * STILE;

    v16bf af[2], bfr[4];
    af[0] = load_frag(&Ab[(wm + lm) * 32 + kb]);
    af[1] = load_frag(&Ab[(wm + 16 + lm) * 32 + kb]);
    #pragma unroll
    for (int tn = 0; tn < 4; ++tn)
      bfr[tn] = load_frag(&Bb[(wn + tn * 16 + lm) * 32 + kb]);

    #pragma unroll
    for (int tm = 0; tm < 2; ++tm)
      #pragma unroll
      for (int tn = 0; tn < 4; ++tn)
        acc[tm][tn] = __builtin_amdgcn_wmma_f32_16x16x32_bf16(
            false, af[tm], false, bfr[tn], (short)0, acc[tm][tn], false, false);

    buf ^= 1;
  }
}

// ---------------- QKV projection ----------------
// qkv = x @ W_attn + b_attn ; scatter into head-major Q (x0.125), K[key][d], V^T[d][key]
__global__ void __launch_bounds__(256)
k_qkv(const u16t* __restrict__ xb, const u16t* __restrict__ Wat,
      const float* __restrict__ b_attn,
      u16t* __restrict__ Qh, u16t* __restrict__ Kh, u16t* __restrict__ Vt) {
  __shared__ u16t As[2 * STILE];
  __shared__ u16t Bs[2 * STILE];
  int m0 = blockIdx.x * 128, n0 = blockIdx.y * 128;
  v8f acc[2][4] = {};
  gemm_main<E_>(xb, Wat, m0, n0, As, Bs, acc);

  int lane = threadIdx.x & 31, wid = threadIdx.x >> 5;
  int wm = (wid & 3) * 32, wn = (wid >> 2) * 64;
  int lm = lane & 15, hi = (lane >> 4) * 8;

  #pragma unroll
  for (int tm = 0; tm < 2; ++tm)
    #pragma unroll
    for (int tn = 0; tn < 4; ++tn)
      #pragma unroll
      for (int r = 0; r < 8; ++r) {
        int m = m0 + wm + tm * 16 + hi + r;
        int n = n0 + wn + tn * 16 + lm;
        float v = acc[tm][tn][r] + b_attn[n];
        int which = n >> 10;            // 0:Q 1:K 2:V
        int e = n & 1023;
        int h = e >> 6, d = e & 63;
        int b = m >> 11, s = m & (S_ - 1);
        size_t bh = (size_t)b * H_ + h;
        if (which == 0)      Qh[(bh * S_ + s) * D_ + d] = f2bf(v * 0.125f); // 1/sqrt(64)
        else if (which == 1) Kh[(bh * S_ + s) * D_ + d] = f2bf(v);
        else                 Vt[(bh * D_ + d) * S_ + s] = f2bf(v);
      }
}

// ---------------- flash attention ----------------
// One wave per (b, h, 16-row q-tile); 32-key blocks; online softmax; WMMA everywhere.
__global__ void __launch_bounds__(128)
k_attn(const u16t* __restrict__ Qh, const u16t* __restrict__ Kh,
       const u16t* __restrict__ Vt, u16t* __restrict__ AttO) {
  __shared__ u16t Plds[4][16 * 32];       // per-wave P bounce buffer (C-layout -> A-layout)
  int wid = threadIdx.x >> 5, lane = threadIdx.x & 31;
  u16t* P = &Plds[wid][0];

  int gw = blockIdx.x * 4 + wid;
  int qt = gw & 127;                       // S/16 = 128
  int h  = (gw >> 7) & 15;
  int b  = gw >> 11;
  int q0 = qt * 16;
  size_t bh = (size_t)b * H_ + h;
  const u16t* Qb = Qh + bh * S_ * D_;
  const u16t* Kb = Kh + bh * S_ * D_;
  const u16t* Vb = Vt + bh * D_ * S_;

  int lm = lane & 15;
  int kb = (lane < 16) ? 0 : 8;
  int hi = (lane >> 4) * 8;

  v16bf qf[2];
  qf[0] = load_frag(&Qb[(q0 + lm) * D_ + 0 * 32 + kb]);
  qf[1] = load_frag(&Qb[(q0 + lm) * D_ + 1 * 32 + kb]);

  v8f O[4] = {};
  float mrow[8], lrow[8];
  #pragma unroll
  for (int r = 0; r < 8; ++r) { mrow[r] = -3.0e38f; lrow[r] = 0.0f; }

  int kend = q0 + 16;                      // number of valid keys (causal)
  for (int k0 = 0; k0 < kend; k0 += 32) {
    // S = Q . K^T for two 16-key subtiles
    v8f sc[2] = {};
    #pragma unroll
    for (int sub = 0; sub < 2; ++sub)
      #pragma unroll
      for (int c = 0; c < 2; ++c) {
        v16bf kf = load_frag(&Kb[(k0 + sub * 16 + lm) * D_ + c * 32 + kb]);
        sc[sub] = __builtin_amdgcn_wmma_f32_16x16x32_bf16(
            false, qf[c], false, kf, (short)0, sc[sub], false, false);
      }

    // causal mask (only diagonal blocks; wave-uniform branch)
    if (k0 + 31 > q0) {
      #pragma unroll
      for (int sub = 0; sub < 2; ++sub)
        #pragma unroll
        for (int r = 0; r < 8; ++r) {
          int row = q0 + hi + r;
          int col = k0 + sub * 16 + lm;
          sc[sub][r] = (col > row) ? -3.0e38f : sc[sub][r];
        }
    }

    // online softmax; row X lives across 16 lanes of one half-wave, VGPR r
    #pragma unroll
    for (int r = 0; r < 8; ++r) {
      float mx = fmaxf(sc[0][r], sc[1][r]);
      #pragma unroll
      for (int off = 8; off >= 1; off >>= 1)
        mx = fmaxf(mx, __shfl_xor(mx, off, 32));
      float mnew  = fmaxf(mrow[r], mx);
      float p0    = __expf(sc[0][r] - mnew);
      float p1    = __expf(sc[1][r] - mnew);
      float alpha = __expf(mrow[r] - mnew);
      float rs = p0 + p1;
      #pragma unroll
      for (int off = 8; off >= 1; off >>= 1)
        rs += __shfl_xor(rs, off, 32);
      lrow[r] = lrow[r] * alpha + rs;
      mrow[r] = mnew;
      #pragma unroll
      for (int dt = 0; dt < 4; ++dt) O[dt][r] *= alpha;
      // C-layout -> LDS (row-major 16x32 P tile)
      P[(hi + r) * 32 + lm]      = f2bf(p0);
      P[(hi + r) * 32 + 16 + lm] = f2bf(p1);
    }

    // reload P as A-fragment; V^T rows give contiguous B-fragments
    v16bf pf = load_frag(&P[lm * 32 + kb]);
    #pragma unroll
    for (int dt = 0; dt < 4; ++dt) {
      v16bf vf = load_frag(&Vb[(dt * 16 + lm) * S_ + k0 + kb]);
      O[dt] = __builtin_amdgcn_wmma_f32_16x16x32_bf16(
          false, pf, false, vf, (short)0, O[dt], false, false);
    }
  }

  // finalize: O /= l, write bf16 [B,S,E]
  #pragma unroll
  for (int r = 0; r < 8; ++r) {
    float inv = 1.0f / lrow[r];
    int row = q0 + hi + r;
    size_t base = ((size_t)b * S_ + row) * E_ + h * D_;
    #pragma unroll
    for (int dt = 0; dt < 4; ++dt)
      AttO[base + dt * 16 + lm] = f2bf(O[dt][r] * inv);
  }
}

// ---------------- output projection ----------------
__global__ void __launch_bounds__(256)
k_proj(const u16t* __restrict__ AttO, const u16t* __restrict__ Wpt,
       const float* __restrict__ b_proj, float* __restrict__ out) {
  __shared__ u16t As[2 * STILE];
  __shared__ u16t Bs[2 * STILE];
  int m0 = blockIdx.x * 128, n0 = blockIdx.y * 128;
  v8f acc[2][4] = {};
  gemm_main<E_>(AttO, Wpt, m0, n0, As, Bs, acc);

  int lane = threadIdx.x & 31, wid = threadIdx.x >> 5;
  int wm = (wid & 3) * 32, wn = (wid >> 2) * 64;
  int lm = lane & 15, hi = (lane >> 4) * 8;

  #pragma unroll
  for (int tm = 0; tm < 2; ++tm)
    #pragma unroll
    for (int tn = 0; tn < 4; ++tn)
      #pragma unroll
      for (int r = 0; r < 8; ++r) {
        int m = m0 + wm + tm * 16 + hi + r;
        int n = n0 + wn + tn * 16 + lm;
        out[(size_t)m * E_ + n] = acc[tm][tn][r] + b_proj[n];
      }
}

extern "C" void kernel_launch(void* const* d_in, const int* in_sizes, int n_in,
                              void* d_out, int out_size, void* d_ws, size_t ws_size,
                              hipStream_t stream) {
  const float* x      = (const float*)d_in[0];
  const float* W_attn = (const float*)d_in[1];
  const float* b_attn = (const float*)d_in[2];
  const float* W_proj = (const float*)d_in[3];
  const float* b_proj = (const float*)d_in[4];
  float* out = (float*)d_out;

  // workspace carve-up (bf16 buffers), total 48 MB
  u16t* xb   = (u16t*)d_ws;                       // [M, E]
  u16t* Wat  = xb   + (size_t)M_ * E_;            // [3E, E]  (W_attn^T)
  u16t* Wpt  = Wat  + (size_t)N3_ * E_;           // [E, E]   (W_proj^T)
  u16t* Qh   = Wpt  + (size_t)E_ * E_;            // [B,H,S,D], pre-scaled
  u16t* Kh   = Qh   + (size_t)B_ * H_ * S_ * D_;  // [B,H,S,D]
  u16t* Vt   = Kh   + (size_t)B_ * H_ * S_ * D_;  // [B,H,D,S] (V^T)
  u16t* AttO = Vt   + (size_t)B_ * H_ * S_ * D_;  // [B,S,E]

  k_cvt<<<(M_ * E_) / 256, 256, 0, stream>>>(x, xb, M_ * E_);
  k_cvt_t<<<dim3(E_ / 256, N3_), 256, 0, stream>>>(W_attn, Wat, E_, N3_);
  k_cvt_t<<<dim3(E_ / 256, E_),  256, 0, stream>>>(W_proj, Wpt, E_, E_);

  k_qkv<<<dim3(M_ / 128, N3_ / 128), 256, 0, stream>>>(xb, Wat, b_attn, Qh, Kh, Vt);

  k_attn<<<(B_ * H_ * (S_ / 16)) / 4, 128, 0, stream>>>(Qh, Kh, Vt, AttO);

  k_proj<<<dim3(M_ / 128, E_ / 128), 256, 0, stream>>>(AttO, Wpt, b_proj, out);
}
